// InAttention_18184891532047
// MI455X (gfx1250) — matrix operand order repaired and checked by reading, not compiled, not run on hardware
//
#include <hip/hip_runtime.h>
#include <hip/hip_bf16.h>

// ---------------------------------------------------------------------------
// MI455X (gfx1250) implementation.
// Compute-bound workload (~193 GFLOP vs ~0.4GB traffic) -> all GEMMs through
// v_wmma_f32_16x16x32_bf16 (fp32 accumulate), wave32, 8 waves/block.
// Data movement: Tensor Data Mover (tensor_load_to_lds) for bf16 tiles with
// descriptor-side LDS row padding; manual path for f32->bf16 converting loads.
// ---------------------------------------------------------------------------

typedef __attribute__((ext_vector_type(16))) __bf16 v16bf;
typedef __attribute__((ext_vector_type(8)))  float  v8f;
typedef unsigned int u32x4 __attribute__((ext_vector_type(4)));
typedef int          i32x8 __attribute__((ext_vector_type(8)));
typedef int          i32x4 __attribute__((ext_vector_type(4)));

#if defined(__has_builtin)
#if __has_builtin(__builtin_amdgcn_tensor_load_to_lds) && \
    __has_builtin(__builtin_amdgcn_s_wait_tensorcnt)
#define HAVE_TDM 1
#endif
#endif
#ifndef HAVE_TDM
#define HAVE_TDM 0
#endif

union AFragU { v16bf v; unsigned int u[8]; };

__device__ __forceinline__ unsigned short f2bf(float f) {
  union { float f; unsigned int u; } x;
  x.f = f;
  unsigned int u = x.u;
  u += 0x7FFFu + ((u >> 16) & 1u);   // round-to-nearest-even
  return (unsigned short)(u >> 16);
}

__device__ __forceinline__ v8f wmma_bf16(v16bf a, v16bf b, v8f c) {
  // D = A(16x32) * B(32x16) + C, fp32 accumulate
  return __builtin_amdgcn_wmma_f32_16x16x32_bf16(false, a, false, b,
                                                 (short)0, c, false, false);
}

#if HAVE_TDM
// Issue a TDM 2-D tile load (bf16 elements) global -> LDS.
// tileW elems/row (row-contiguous in memory, rows strideElems apart),
// tileH rows. LDS receives rows padded: after every 2^(padInt+1) DWORDs
// stored, skip (padAmt+1) DWORDs -> LDS row stride control.
// D# layout per CDNA5 ISA. One DMA per call (wave op).
__device__ __forceinline__ void tdm_load_2d_bf16(
    const unsigned short* gptr, unsigned ldsByteAddr,
    unsigned tileW, unsigned tileH, unsigned strideElems,
    unsigned padInt, unsigned padAmt) {
  unsigned long long ga = (unsigned long long)gptr;
  u32x4 g0;
  g0[0] = 1u;                                          // count=1, user desc
  g0[1] = ldsByteAddr;                                 // [63:32] lds_addr
  g0[2] = (unsigned)(ga & 0xFFFFFFFFu);                // [95:64] addr lo
  g0[3] = (unsigned)((ga >> 32) & 0x1FFFFFFu) | (2u << 30); // addr hi | type=2
  i32x8 g1;
  g1[0] = (int)((1u << 16)            // data_size = 2 bytes
              | (1u << 20)            // pad_enable
              | (padInt << 22)        // pad_interval
              | (padAmt << 25));      // pad_amount
  g1[1] = (int)((tileW & 0xFFFFu) << 16);              // tensor_dim0[15:0]
  g1[2] = (int)(((tileW >> 16) & 0xFFFFu) | ((tileH & 0xFFFFu) << 16));
  g1[3] = (int)(((tileH >> 16) & 0xFFFFu) | ((tileW & 0xFFFFu) << 16)); // tile_dim0
  g1[4] = (int)(tileH & 0xFFFFu);                      // tile_dim1 (tile_dim2=0)
  g1[5] = (int)strideElems;                            // tensor_dim0_stride lo
  g1[6] = 0;                                           // stride hi / dim1_stride
  g1[7] = 0;
  i32x4 z4 = {0, 0, 0, 0};
#if __clang_major__ >= 23
  i32x8 z8 = {0, 0, 0, 0, 0, 0, 0, 0};
  __builtin_amdgcn_tensor_load_to_lds(g0, g1, z4, z4, z8, 0);
#else
  __builtin_amdgcn_tensor_load_to_lds(g0, g1, z4, z4, 0);
#endif
}
#endif  // HAVE_TDM

// A fragment (16x32 bf16) from LDS row-major [row][k], row stride ld (ushorts).
// lane%16 = M; lanes 0-15 hold K 0-15, lanes 16-31 K 16-31 (contiguous k, same as B).
__device__ __forceinline__ v16bf load_a_frag(const unsigned short* lds,
                                             int row_base, int ld, int k0) {
  const int lane = threadIdx.x & 31;
  const int half = lane >> 4;
  const unsigned short* p = lds + (row_base + (lane & 15)) * ld;
  AFragU f;
#pragma unroll
  for (int i = 0; i < 8; ++i) {
    int k = k0 + half * 16 + 2 * i;
    f.u[i] = *(const unsigned int*)(p + k);
  }
  return f.v;
}

// B fragment (32x16 bf16) from LDS stored K-contiguous per column: [n][k].
__device__ __forceinline__ v16bf load_b_frag(const unsigned short* lds,
                                             int col_base, int ld, int k0) {
  const int lane = threadIdx.x & 31;
  const int half = lane >> 4;
  const unsigned short* p = lds + (col_base + (lane & 15)) * ld;
  AFragU f;
#pragma unroll
  for (int i = 0; i < 8; ++i) {
    int k = k0 + half * 16 + 2 * i;
    f.u[i] = *(const unsigned int*)(p + k);
  }
  return f.v;
}

// ---------------------------------------------------------------------------
// Generic batched GEMM: C[z] = act(A[z] * B + bias).  A: MxK (f32 or bf16),
// B: KxN f32 weights (bf16-converted in-kernel; stay hot in 192MB L2),
// C: MxN (f32 or bf16). All call sites: M%128==0, N%128==0, K%32==0.
// bf16-A tiles arrive via TDM (double buffered); f32-A converted manually.
// ---------------------------------------------------------------------------
#define GBM 128
#define GBN 128
#define GBK 32
#define GLD 40   // LDS row stride (ushorts): 80B, 16B-aligned, conflict-free

__global__ __launch_bounds__(256) void gemm_kernel(
    const void* __restrict__ Aptr, int a_bf16, long long aBatch,
    const float* __restrict__ Bmat,
    void* __restrict__ Cptr, int c_bf16, long long cBatch,
    int M, int N, int K, const float* __restrict__ bias, int relu) {
  __shared__ __align__(16) unsigned short Al[2 * GBM * GLD];
  __shared__ __align__(16) unsigned short Bt[GBN * GLD];
  const int tid  = threadIdx.x;
  const int wid  = tid >> 5;
  const int wm   = wid & 3;          // 4 wave rows  (32 rows each)
  const int wn   = wid >> 2;         // 2 wave cols  (64 cols each)
  const int lane = tid & 31, half = lane >> 4, nl = lane & 15;
  const int z  = blockIdx.z;
  const int m0 = blockIdx.y * GBM;
  const int n0 = blockIdx.x * GBN;

  const float*          Af = (const float*)Aptr + (size_t)z * aBatch;
  const unsigned short* Ab = (const unsigned short*)Aptr + (size_t)z * aBatch;

  v8f zero = {0.f, 0.f, 0.f, 0.f, 0.f, 0.f, 0.f, 0.f};
  v8f acc[2][4];
#pragma unroll
  for (int mi = 0; mi < 2; ++mi)
#pragma unroll
    for (int ni = 0; ni < 4; ++ni) acc[mi][ni] = zero;

#if HAVE_TDM
  const unsigned AlByte = (unsigned)(size_t)(void*)&Al[0];
  if (a_bf16 && wid == 0)   // prologue: DMA first A tile into buffer 0
    tdm_load_2d_bf16(Ab + (size_t)m0 * K, AlByte, GBK, GBM, (unsigned)K, 3, 3);
#endif

  int cur = 0;
  for (int kt = 0; kt < K; kt += GBK) {
#pragma unroll
    for (int i = 0; i < 4; ++i) {                // B tile, store transposed
      int idx = tid + i * 256;
      int kr  = idx >> 5;
      int c4  = (idx & 31) << 2;
      float4 v = *(const float4*)(Bmat + (size_t)(kt + kr) * N + n0 + c4);
      Bt[(c4 + 0) * GLD + kr] = f2bf(v.x);
      Bt[(c4 + 1) * GLD + kr] = f2bf(v.y);
      Bt[(c4 + 2) * GLD + kr] = f2bf(v.z);
      Bt[(c4 + 3) * GLD + kr] = f2bf(v.w);
    }
    if (a_bf16) {
#if HAVE_TDM
      if (wid == 0) {                            // DMA next tile, wait current
        if (kt + GBK < K) {
          tdm_load_2d_bf16(Ab + (size_t)m0 * K + kt + GBK,
                           AlByte + (unsigned)((cur ^ 1) * GBM * GLD * 2),
                           GBK, GBM, (unsigned)K, 3, 3);
          __builtin_amdgcn_s_wait_tensorcnt(1);  // in-order: current done
        } else {
          __builtin_amdgcn_s_wait_tensorcnt(0);
        }
      }
#else
#pragma unroll
      for (int i = 0; i < 2; ++i) {              // 512 x uint4 (8 bf16 each)
        int idx = tid + i * 256;
        int row = idx >> 2;
        int c8  = (idx & 3) << 3;
        uint4 v = *(const uint4*)(Ab + (size_t)(m0 + row) * K + kt + c8);
        *(uint4*)&Al[row * GLD + c8] = v;
      }
#endif
    } else {
#pragma unroll
      for (int i = 0; i < 4; ++i) {              // 1024 x float4, cvt to bf16
        int idx = tid + i * 256;
        int row = idx >> 3;
        int c4  = (idx & 7) << 2;
        float4 v = *(const float4*)(Af + (size_t)(m0 + row) * K + kt + c4);
        unsigned int* d = (unsigned int*)&Al[row * GLD + c4];
        d[0] = (unsigned int)f2bf(v.x) | ((unsigned int)f2bf(v.y) << 16);
        d[1] = (unsigned int)f2bf(v.z) | ((unsigned int)f2bf(v.w) << 16);
      }
    }
    __syncthreads();

#if HAVE_TDM
    const unsigned short* Acur = Al + (a_bf16 ? cur * (GBM * GLD) : 0);
#else
    const unsigned short* Acur = Al;
#endif
    v16bf afr[2];
#pragma unroll
    for (int mi = 0; mi < 2; ++mi)
      afr[mi] = load_a_frag(Acur, wm * 32 + mi * 16, GLD, 0);
#pragma unroll
    for (int ni = 0; ni < 4; ++ni) {
      v16bf bfr = load_b_frag(Bt, wn * 64 + ni * 16, GLD, 0);
#pragma unroll
      for (int mi = 0; mi < 2; ++mi)
        acc[mi][ni] = wmma_bf16(afr[mi], bfr, acc[mi][ni]);
    }
    __syncthreads();
    cur ^= 1;
  }

#pragma unroll
  for (int mi = 0; mi < 2; ++mi)
#pragma unroll
    for (int ni = 0; ni < 4; ++ni) {
      int col  = n0 + wn * 64 + ni * 16 + nl;
      float bv = bias ? bias[col] : 0.0f;
#pragma unroll
      for (int r = 0; r < 8; ++r) {
        int row = m0 + wm * 32 + mi * 16 + half * 8 + r;
        float v = acc[mi][ni][r] + bv;
        if (relu) v = fmaxf(v, 0.0f);
        size_t off = (size_t)z * cBatch + (size_t)row * N + col;
        if (c_bf16) ((unsigned short*)Cptr)[off] = f2bf(v);
        else        ((float*)Cptr)[off] = v;
      }
    }
}

// ---------------------------------------------------------------------------
// Flash attention: 64 queries x 1 head per block, streams 64-key chunks with
// online softmax. Q/K/V/O bf16, row stride 1024 (head h = cols 64h..64h+63).
// Q and K tiles DMA'd by the TDM; V loaded manually (needs transpose).
// Waves: wm in {0,1} (32 query rows), wn in {0..3} (16 keys / 16 out dims).
// ---------------------------------------------------------------------------
#define FLD 72   // 144B row stride: 16B aligned, conflict-free (36 banks)

__global__ __launch_bounds__(256) void flash_kernel(
    const unsigned short* __restrict__ Qb, int qStartRow,
    const unsigned short* __restrict__ Kb, const unsigned short* __restrict__ Vb,
    int kvBatchRows, unsigned short* __restrict__ Ob,
    int Skv, int causal, float scale) {
  __shared__ __align__(16) unsigned short Ql[64 * FLD];
  __shared__ __align__(16) unsigned short Kl[64 * FLD];  // [key][d]
  __shared__ __align__(16) unsigned short Vt[64 * FLD];  // [d][key]
  __shared__ __align__(16) unsigned short Pl[64 * FLD];  // [qrow][key]
  __shared__ float redM[4][64];
  __shared__ float redS[4][64];

  const int tid  = threadIdx.x;
  const int wid  = tid >> 5;
  const int wn   = wid & 3;
  const int wm   = wid >> 2;
  const int lane = tid & 31, half = lane >> 4, nl = lane & 15;
  const int h  = blockIdx.y;
  const int z  = blockIdx.z;
  const int qb = blockIdx.x;

  const size_t qbase  = (size_t)z * 16384 + qStartRow + (size_t)qb * 64;
  const size_t kvbase = (size_t)z * kvBatchRows;

#if HAVE_TDM
  if (wid == 0)   // stage Q tile (64x64 bf16) via TDM; 32 data + 4 pad DWORDs
    tdm_load_2d_bf16(Qb + qbase * 1024 + h * 64,
                     (unsigned)(size_t)(void*)&Ql[0], 64, 64, 1024, 4, 3);
#else
#pragma unroll
  for (int i = 0; i < 2; ++i) {
    int idx = tid + i * 256;
    int row = idx >> 3;
    int c8  = (idx & 7) << 3;
    uint4 v = *(const uint4*)(Qb + (qbase + row) * 1024 + h * 64 + c8);
    *(uint4*)&Ql[row * FLD + c8] = v;
  }
#endif

  v8f zero = {0.f, 0.f, 0.f, 0.f, 0.f, 0.f, 0.f, 0.f};
  v8f o[2];
  float Mrun[2][8], Srun[2][8];
#pragma unroll
  for (int mi = 0; mi < 2; ++mi) {
    o[mi] = zero;
#pragma unroll
    for (int r = 0; r < 8; ++r) { Mrun[mi][r] = -1e30f; Srun[mi][r] = 0.f; }
  }

  const int nch = causal ? (qb + 1) : (Skv >> 6);
  for (int c = 0; c < nch; ++c) {
    const int k0 = c << 6;
#if HAVE_TDM
    if (wid == 0)                                // K chunk -> Kl[key][d]
      tdm_load_2d_bf16(Kb + (kvbase + k0) * 1024 + h * 64,
                       (unsigned)(size_t)(void*)&Kl[0], 64, 64, 1024, 4, 3);
#else
#pragma unroll
    for (int i = 0; i < 2; ++i) {
      int idx = tid + i * 256;
      int row = idx >> 3;
      int c8  = (idx & 7) << 3;
      uint4 v = *(const uint4*)(Kb + (kvbase + k0 + row) * 1024 + h * 64 + c8);
      *(uint4*)&Kl[row * FLD + c8] = v;
    }
#endif
#pragma unroll
    for (int i = 0; i < 16; ++i) {               // V chunk -> Vt[d][key]
      int idx = tid + i * 256;
      int kr = idx >> 6, d = idx & 63;
      Vt[d * FLD + kr] = Vb[(kvbase + k0 + kr) * 1024 + h * 64 + d];
    }
#if HAVE_TDM
    if (wid == 0) __builtin_amdgcn_s_wait_tensorcnt(0);  // Q (chunk 0) + K
#endif
    __syncthreads();

    v8f s[2]; s[0] = zero; s[1] = zero;          // scores: S = Q * K^T
#pragma unroll
    for (int kk = 0; kk < 64; kk += 32) {
      v16bf bfr = load_b_frag(Kl, wn * 16, FLD, kk);
#pragma unroll
      for (int mi = 0; mi < 2; ++mi) {
        v16bf afr = load_a_frag(Ql, wm * 32 + mi * 16, FLD, kk);
        s[mi] = wmma_bf16(afr, bfr, s[mi]);
      }
    }

    float ps[2][8];
    const int keyA = k0 + wn * 16 + nl;
#pragma unroll
    for (int mi = 0; mi < 2; ++mi)
#pragma unroll
      for (int r = 0; r < 8; ++r) {
        float v = s[mi][r] * scale;
        if (causal) {
          int qrow = qb * 64 + wm * 32 + mi * 16 + half * 8 + r;
          if (keyA > qrow) v = -1e30f;
        }
        ps[mi][r] = v;
      }

    float rm[2][8];                              // row max over this wave's keys
#pragma unroll
    for (int mi = 0; mi < 2; ++mi)
#pragma unroll
      for (int r = 0; r < 8; ++r) rm[mi][r] = ps[mi][r];
#pragma unroll
    for (int off = 1; off < 16; off <<= 1)
#pragma unroll
      for (int mi = 0; mi < 2; ++mi)
#pragma unroll
        for (int r = 0; r < 8; ++r)
          rm[mi][r] = fmaxf(rm[mi][r], __shfl_xor(rm[mi][r], off, 32));
    if (nl == 0) {
#pragma unroll
      for (int mi = 0; mi < 2; ++mi)
#pragma unroll
        for (int r = 0; r < 8; ++r)
          redM[wn][wm * 32 + mi * 16 + half * 8 + r] = rm[mi][r];
    }
    __syncthreads();

    float alpha[2][8];
#pragma unroll
    for (int mi = 0; mi < 2; ++mi)
#pragma unroll
      for (int r = 0; r < 8; ++r) {
        int row = wm * 32 + mi * 16 + half * 8 + r;
        float mc = fmaxf(fmaxf(redM[0][row], redM[1][row]),
                         fmaxf(redM[2][row], redM[3][row]));
        float mn = fmaxf(Mrun[mi][r], mc);
        alpha[mi][r] = __expf(Mrun[mi][r] - mn);
        Mrun[mi][r] = mn;
      }

    float rs[2][8];                              // P = exp(s - Mnew), row sums
#pragma unroll
    for (int mi = 0; mi < 2; ++mi)
#pragma unroll
      for (int r = 0; r < 8; ++r) {
        float p = __expf(ps[mi][r] - Mrun[mi][r]);
        ps[mi][r] = p;
        rs[mi][r] = p;
      }
#pragma unroll
    for (int off = 1; off < 16; off <<= 1)
#pragma unroll
      for (int mi = 0; mi < 2; ++mi)
#pragma unroll
        for (int r = 0; r < 8; ++r)
          rs[mi][r] += __shfl_xor(rs[mi][r], off, 32);
    if (nl == 0) {
#pragma unroll
      for (int mi = 0; mi < 2; ++mi)
#pragma unroll
        for (int r = 0; r < 8; ++r)
          redS[wn][wm * 32 + mi * 16 + half * 8 + r] = rs[mi][r];
    }
#pragma unroll
    for (int mi = 0; mi < 2; ++mi)               // P -> LDS (C-layout -> A-layout)
#pragma unroll
      for (int r = 0; r < 8; ++r)
        Pl[(wm * 32 + mi * 16 + half * 8 + r) * FLD + wn * 16 + nl] =
            f2bf(ps[mi][r]);
    __syncthreads();

#pragma unroll
    for (int mi = 0; mi < 2; ++mi)
#pragma unroll
      for (int r = 0; r < 8; ++r) {
        int row = wm * 32 + mi * 16 + half * 8 + r;
        float sc = redS[0][row] + redS[1][row] + redS[2][row] + redS[3][row];
        Srun[mi][r] = Srun[mi][r] * alpha[mi][r] + sc;
        o[mi][r] = o[mi][r] * alpha[mi][r];
      }

#pragma unroll
    for (int kk = 0; kk < 64; kk += 32) {        // O += P * V
      v16bf bfr = load_b_frag(Vt, wn * 16, FLD, kk);
#pragma unroll
      for (int mi = 0; mi < 2; ++mi) {
        v16bf afr = load_a_frag(Pl, wm * 32 + mi * 16, FLD, kk);
        o[mi] = wmma_bf16(afr, bfr, o[mi]);
      }
    }
    __syncthreads();
  }

#pragma unroll
  for (int mi = 0; mi < 2; ++mi)
#pragma unroll
    for (int r = 0; r < 8; ++r) {
      int rowL = wm * 32 + mi * 16 + half * 8 + r;
      float v = o[mi][r] / Srun[mi][r];
      Ob[(qbase + rowL) * 1024 + h * 64 + wn * 16 + nl] = f2bf(v);
    }
}

// Mean-pool groups of 4 consecutive rows: (B,1024,1024) -> (B,256,1024)
__global__ __launch_bounds__(256) void pool_kernel(const float* __restrict__ src,
                                                   float* __restrict__ dst) {
  int idx = blockIdx.x * 256 + threadIdx.x;      // exactly 2*256*1024 threads
  int d = idx & 1023;
  int p = (idx >> 10) & 255;
  int z = idx >> 18;
  const float* s = src + ((size_t)z * 1024 + (size_t)p * 4) * 1024 + d;
  dst[idx] = 0.25f * (s[0] + s[1024] + s[2048] + s[3072]);
}

// ---------------------------------------------------------------------------
extern "C" void kernel_launch(void* const* d_in, const int* in_sizes, int n_in,
                              void* d_out, int out_size, void* d_ws,
                              size_t ws_size, hipStream_t stream) {
  (void)in_sizes; (void)n_in; (void)out_size; (void)ws_size;
  const float* hidden = (const float*)d_in[0];
  const float* Wq = (const float*)d_in[1];
  const float* Wk = (const float*)d_in[2];
  const float* Wv = (const float*)d_in[3];
  const float* Wo = (const float*)d_in[4];
  const float* W1 = (const float*)d_in[5];
  const float* b1 = (const float*)d_in[6];
  const float* W2 = (const float*)d_in[7];
  const float* b2 = (const float*)d_in[8];

  char* ws = (char*)d_ws;
  size_t off = 0;
  auto carve = [&](size_t bytes) -> void* {
    void* p = ws + off;
    off += (bytes + 255) & ~(size_t)255;
    return p;
  };
  unsigned short* Qb   = (unsigned short*)carve(2ull * 16384 * 1024 * 2);
  unsigned short* Obuf = (unsigned short*)carve(2ull * 16384 * 1024 * 2);
  unsigned short* Kini = (unsigned short*)carve(2ull * 1024 * 1024 * 2);
  unsigned short* Vini = (unsigned short*)carve(2ull * 1024 * 1024 * 2);
  unsigned short* Kcmp = (unsigned short*)carve(2ull * 256 * 1024 * 2);
  unsigned short* Vcmp = (unsigned short*)carve(2ull * 256 * 1024 * 2);
  float* H1    = (float*)carve(2ull * 1024 * 512 * 4);
  float* Cfull = (float*)carve(2ull * 1024 * 1024 * 4);
  float* Cpool = (float*)carve(2ull * 256 * 1024 * 4);

  const dim3 blk(256);
  const long long SB = 16384LL * 1024;   // full-sequence batch stride (elems)

  // Q = hidden @ Wq (all 16384 rows; both branches share Wq)   -> bf16
  gemm_kernel<<<dim3(8, 128, 2), blk, 0, stream>>>(hidden, 0, SB, Wq,
      Qb, 1, SB, 16384, 1024, 1024, nullptr, 0);
  // K/V of initial 1024 tokens -> bf16
  gemm_kernel<<<dim3(8, 8, 2), blk, 0, stream>>>(hidden, 0, SB, Wk,
      Kini, 1, 1024LL * 1024, 1024, 1024, 1024, nullptr, 0);
  gemm_kernel<<<dim3(8, 8, 2), blk, 0, stream>>>(hidden, 0, SB, Wv,
      Vini, 1, 1024LL * 1024, 1024, 1024, 1024, nullptr, 0);
  // MLP: H1 = relu(initial @ W1 + b1);  Cfull = H1 @ W2 + b2
  gemm_kernel<<<dim3(4, 8, 2), blk, 0, stream>>>(hidden, 0, SB, W1,
      H1, 0, 1024LL * 512, 1024, 512, 1024, b1, 1);
  gemm_kernel<<<dim3(8, 8, 2), blk, 0, stream>>>(H1, 0, 1024LL * 512, W2,
      Cfull, 0, 1024LL * 1024, 1024, 1024, 512, b2, 0);
  pool_kernel<<<dim3(2048), blk, 0, stream>>>(Cfull, Cpool);
  // K/V of pooled context (256 tokens) -> bf16
  gemm_kernel<<<dim3(8, 2, 2), blk, 0, stream>>>(Cpool, 0, 256LL * 1024, Wk,
      Kcmp, 1, 256LL * 1024, 256, 1024, 1024, nullptr, 0);
  gemm_kernel<<<dim3(8, 2, 2), blk, 0, stream>>>(Cpool, 0, 256LL * 1024, Wv,
      Vcmp, 1, 256LL * 1024, 256, 1024, 1024, nullptr, 0);
  // Attention core (scale = 1/sqrt(64))
  flash_kernel<<<dim3(16, 16, 2), blk, 0, stream>>>(Qb, 0, Kini, Vini, 1024,
      Obuf, 1024, 1, 0.125f);                      // causal self-attn, rows 0..1023
  flash_kernel<<<dim3(240, 16, 2), blk, 0, stream>>>(Qb, 1024, Kcmp, Vcmp, 256,
      Obuf, 256, 0, 0.125f);                       // cross-attn, rows 1024..16383
  // Final projection straight into d_out (fp32, reference concat order)
  gemm_kernel<<<dim3(8, 128, 2), blk, 0, stream>>>(Obuf, 1, SB, Wo,
      d_out, 0, SB, 16384, 1024, 1024, nullptr, 0);
}
